// Attention_42494406426929
// MI455X (gfx1250) — matrix operand order repaired and checked by reading, not compiled, not run on hardware
//
#include <hip/hip_runtime.h>
#include <hip/hip_bf16.h>

typedef __attribute__((ext_vector_type(16))) _Float16 v16h;
typedef __attribute__((ext_vector_type(8)))  _Float16 v8h;
typedef __attribute__((ext_vector_type(8)))  float    v8f;

union HF { v16h v; v8h h8[2]; };

__device__ __forceinline__ v8f wmma_f16(v16h a, v16h b, v8f c) {
    return __builtin_amdgcn_wmma_f32_16x16x32_f16(false, a, false, b, (short)0, c, false, false);
}
__device__ __forceinline__ float rowmax16(float v) {
    #pragma unroll
    for (int off = 8; off >= 1; off >>= 1) v = fmaxf(v, __shfl_xor(v, off, 16));
    return v;
}
__device__ __forceinline__ float rowsum16(float v) {
    #pragma unroll
    for (int off = 8; off >= 1; off >>= 1) v += __shfl_xor(v, off, 16);
    return v;
}
// CDNA5 async global->LDS copy (16B per lane), tracked by ASYNCcnt.  The LDS
// aperture maps addr[31:0] to the LDS offset, so the low 32 bits of a generic
// pointer into __shared__ are the LDS byte address.
__device__ __forceinline__ void async_cp16(void* lds, const void* gaddr) {
    unsigned l32 = (unsigned)(unsigned long long)lds;
    asm volatile("global_load_async_to_lds_b128 %0, %1, off"
                 :: "v"(l32), "v"(gaddr) : "memory");
}
// Wait until at most N async ops remain in flight (in-order drain).
#define ASYNC_WAIT(n) asm volatile("s_wait_asynccnt " #n ::: "memory")

// ---------------------------------------------------------------------------
// One-time converts: x fp32 -> f16, W[K][N] fp32 -> WT[N][K] f16, V -> V^T.
// ---------------------------------------------------------------------------
__global__ __launch_bounds__(256) void cvt_f16_kernel(const float* __restrict__ in,
                                                      _Float16* __restrict__ out) {
    const size_t i = ((size_t)blockIdx.x * 256 + threadIdx.x) * 8;
    #pragma unroll
    for (int j = 0; j < 8; j++) out[i + j] = (_Float16)in[i + j];
}

__global__ __launch_bounds__(256) void wt_transpose_kernel(const float* __restrict__ W,
                                                           _Float16* __restrict__ WT,
                                                           int K, int N) {
    __shared__ _Float16 L[32][33];
    const int nb = blockIdx.x * 32, kb = blockIdx.y * 32;
    const int a = threadIdx.x & 31, q = threadIdx.x >> 5;
    #pragma unroll
    for (int i = 0; i < 4; i++) {          // read: consecutive tid -> consecutive N
        const int k = q + 8 * i;
        L[a][k] = (_Float16)W[(size_t)(kb + k) * N + nb + a];
    }
    __syncthreads();
    #pragma unroll
    for (int i = 0; i < 4; i++) {          // write: consecutive tid -> consecutive K
        const int n = q + 8 * i;
        WT[(size_t)(nb + n) * K + kb + a] = L[n][a];
    }
}

// V (f16, [b][s][g][128] inside kv) -> vT (f16, [b*4+g][128][2048])
__global__ __launch_bounds__(256) void v_transpose_kernel(const _Float16* __restrict__ kvh,
                                                          _Float16* __restrict__ vT) {
    __shared__ _Float16 L[32][33];
    const int pb = blockIdx.x * 32, db = blockIdx.y * 32, bg = blockIdx.z;
    const int b = bg >> 2, g = bg & 3;
    const int a = threadIdx.x & 31, q = threadIdx.x >> 5;
    #pragma unroll
    for (int i = 0; i < 4; i++) {
        const int p = q + 8 * i;
        L[a][p] = kvh[(size_t)(b * 2048 + pb + p) * 1024 + 512 + g * 128 + db + a];
    }
    __syncthreads();
    #pragma unroll
    for (int i = 0; i < 4; i++) {
        const int d = q + 8 * i;
        vT[((size_t)bg * 128 + db + d) * 2048 + pb + a] = L[d][a];
    }
}

// ---------------------------------------------------------------------------
// GEMM: C[M,N] = A[M,K] @ BT[N,K]^T, all-f16 operands, double-buffered async
// LDS staging.  BM=128, BN=128, BK=32; 8 waves (4x2), wave tile 32x64 = 8 WMMA.
// ---------------------------------------------------------------------------
#define BK 32
#define LDS_S 40   // padded stride (halves), 16B aligned

template<bool OUT_HALF>
__global__ __launch_bounds__(256) void gemm_f16_wmma(
    const _Float16* __restrict__ A, const _Float16* __restrict__ BT,
    void* __restrict__ Cv, int M, int N, int K)
{
    __shared__ __align__(16) _Float16 As[2][128 * LDS_S];
    __shared__ __align__(16) _Float16 Bs[2][128 * LDS_S];

    const int tid = threadIdx.x;
    const int l  = tid & 31, lo = l & 15, hi = l >> 4;
    const int w  = tid >> 5, wm = w >> 1, wn = w & 1;
    const int bm = blockIdx.y * 128, bn = blockIdx.x * 128;

    const _Float16* Arow = A  + (size_t)bm * K;
    const _Float16* Brow = BT + (size_t)bn * K;

    v8f acc[2][4] = {};

    // stage tile k0 into buffer `buf`: 4 async instructions per wave
    auto stage = [&](int buf, int k0) {
        #pragma unroll
        for (int j = 0; j < 2; j++) {
            const int idx = tid * 2 + j;          // 0..511
            const int row = idx >> 2, part = (idx & 3) * 8;
            async_cp16(&As[buf][row * LDS_S + part], Arow + (size_t)row * K + k0 + part);
            async_cp16(&Bs[buf][row * LDS_S + part], Brow + (size_t)row * K + k0 + part);
        }
    };

    stage(0, 0);
    int it = 0;
    for (int k0 = 0; k0 < K; k0 += BK, it ^= 1) {
        if (k0 + BK < K) { stage(it ^ 1, k0 + BK); ASYNC_WAIT(4); }
        else             { ASYNC_WAIT(0); }
        __syncthreads();

        v16h am[2], bf[4];
        #pragma unroll
        for (int i = 0; i < 2; i++) {
            const _Float16* p = &As[it][(wm * 32 + i * 16 + lo) * LDS_S + hi * 8];
            HF u; u.h8[0] = *(const v8h*)p; u.h8[1] = *(const v8h*)(p + 16);
            am[i] = u.v;
        }
        #pragma unroll
        for (int j = 0; j < 4; j++) {
            const _Float16* p = &Bs[it][(wn * 64 + j * 16 + lo) * LDS_S + hi * 8];
            HF u; u.h8[0] = *(const v8h*)p; u.h8[1] = *(const v8h*)(p + 16);
            bf[j] = u.v;
        }
        #pragma unroll
        for (int i = 0; i < 2; i++)
            #pragma unroll
            for (int j = 0; j < 4; j++)
                acc[i][j] = wmma_f16(am[i], bf[j], acc[i][j]);
        __syncthreads();   // all waves done reading buf `it` before it is restaged
    }

    #pragma unroll
    for (int i = 0; i < 2; i++)
        #pragma unroll
        for (int j = 0; j < 4; j++)
            #pragma unroll
            for (int r = 0; r < 8; r++) {
                const int gr = bm + wm * 32 + i * 16 + r + hi * 8;
                const int gc = bn + wn * 64 + j * 16 + lo;
                if (OUT_HALF) ((_Float16*)Cv)[(size_t)gr * N + gc] = (_Float16)acc[i][j][r];
                else          ((float*)Cv)[(size_t)gr * N + gc]    = acc[i][j][r];
            }
}

// ---------------------------------------------------------------------------
// RoPE (in place, fp32 math).
// ---------------------------------------------------------------------------
__global__ __launch_bounds__(256) void rope_kernel(_Float16* base, int heads, int rowStride, int total)
{
    int idx = blockIdx.x * blockDim.x + threadIdx.x;
    if (idx >= total) return;
    const int j  = idx & 63;
    const int t  = idx >> 6;
    const int h  = t % heads;
    const int s_ = (t / heads) & 2047;
    const int b  = t / (heads * 2048);
    _Float16* p = base + (size_t)(b * 2048 + s_) * rowStride + h * 128;
    const float inv = __expf(-(float)j * (9.210340371976184f / 64.0f)); // 10000^(-j/64)
    float sn, cs; __sincosf((float)s_ * inv, &sn, &cs);
    const float x1 = (float)p[j], x2 = (float)p[j + 64];
    p[j]      = (_Float16)(x1 * cs - x2 * sn);
    p[j + 64] = (_Float16)(x2 * cs + x1 * sn);
}

// ---------------------------------------------------------------------------
// Flash attention: grid (S/32, GROUPS, B); block 256 = 8 waves =
// 4 heads x 2 query sub-tiles, all sharing one double-buffered K/V stream.
// 64-key chunks; per wave: 16 query rows, 32 WMMAs per chunk.
// All wave-uniform guards go through readfirstlane so they lower to scalar
// branches (EXEC stays all-ones across every WMMA, per ISA requirement).
// ---------------------------------------------------------------------------
#define SKK 136  // Ks stride: 128 + 8
#define SVV 72   // Vt stride: 64 + 8
#define SPP 72   // P  stride: 64 + 8

__global__ __launch_bounds__(256) void attn_fa_wmma(
    const _Float16* __restrict__ qh, const _Float16* __restrict__ kvh,
    const _Float16* __restrict__ vT, _Float16* __restrict__ ctxh)
{
    __shared__ __align__(16) _Float16 Ks[2][64 * SKK];      // K chunk [key][d]
    __shared__ __align__(16) _Float16 Vt[2][128 * SVV];     // V chunk [d][key]
    __shared__ __align__(16) _Float16 Pw[8 * 16 * SPP];     // per-wave P (16x64)

    const int tid = threadIdx.x;
    const int l = tid & 31, lo = l & 15, hi = l >> 4;
    const int w = tid >> 5;
    const int qt = blockIdx.x, g = blockIdx.y, b = blockIdx.z;
    const int h  = g * 4 + (w & 3);
    const int bg = b * 4 + g;
    const int qbase = qt * 32 + (w >> 2) * 16;
    // wave-uniform copy in an SGPR for scalar control flow
    const int qb_s = __builtin_amdgcn_readfirstlane(qbase);
    const float scale = 0.08838834764831845f;   // 1/sqrt(128)

    v16h qa[4];
    {
        const _Float16* qp = qh + (((size_t)b * 2048 + qbase + lo) * 16 + h) * 128;
        #pragma unroll
        for (int c = 0; c < 4; c++) {
            const _Float16* p = qp + c * 32 + hi * 8;
            HF u; u.h8[0] = *(const v8h*)p; u.h8[1] = *(const v8h*)(p + 16);
            qa[c] = u.v;
        }
    }

    v8f acc[8] = {};
    float m[8], su[8];
    #pragma unroll
    for (int r = 0; r < 8; r++) { m[r] = -3.0e38f; su[r] = 0.f; }

    // stage chunk k0 into buffer `buf`: K 64x128 + Vt 128x64 halves,
    // 2048 16B transfers / 256 threads = 8 async instructions per wave
    const _Float16* kbase = kvh + (size_t)b * 2048 * 1024 + g * 128;
    const _Float16* vbase = vT + (size_t)bg * 128 * 2048;
    auto stage = [&](int buf, int k0) {
        #pragma unroll
        for (int j = 0; j < 4; j++) {
            const int ki = tid * 4 + j;                       // 0..1023
            const int krow = ki >> 4, kpart = (ki & 15) * 8;  // 64 x 16 chunks
            async_cp16(&Ks[buf][krow * SKK + kpart],
                       kbase + (size_t)(k0 + krow) * 1024 + kpart);
            const int vd = ki >> 2, vpart = (ki & 3) * 8;     // 256 x 4 chunks
            async_cp16(&Vt[buf][vd * SVV + vpart],
                       vbase + (size_t)vd * 2048 + k0 + vpart);
        }
    };

    const int myNch  = (qb_s + 16 + 63) >> 6;                 // chunks this wave needs (SGPR)
    const int allNch = (qt * 32 + 16 + 16 + 63) >> 6;         // block-wide max

    stage(0, 0);
    int it = 0;
    for (int c2 = 0; c2 < allNch; c2++, it ^= 1) {
        const int k0 = c2 * 64;
        if (c2 + 1 < allNch) { stage(it ^ 1, (c2 + 1) * 64); ASYNC_WAIT(8); }
        else                 { ASYNC_WAIT(0); }
        __syncthreads();

        if (c2 < myNch) {
            // --- scores: four 16-key subtiles; scalar-branch skip/mask split ---
            v8f sc[4];
            #pragma unroll
            for (int sub = 0; sub < 4; sub++) {
                const int klo = k0 + sub * 16;                // scalar
                if (klo <= qb_s + 15) {
                    v8f s = {};
                    #pragma unroll
                    for (int c = 0; c < 4; c++) {
                        const _Float16* p = &Ks[it][(sub * 16 + lo) * SKK + c * 32 + hi * 8];
                        HF u; u.h8[0] = *(const v8h*)p; u.h8[1] = *(const v8h*)(p + 16);
                        s = wmma_f16(qa[c], u.v, s);
                    }
                    if (klo + 15 <= qb_s) {
                        // fully inside the causal triangle: scale only
                        #pragma unroll
                        for (int r = 0; r < 8; r++) s[r] *= scale;
                    } else {
                        // diagonal subtile: per-element mask
                        #pragma unroll
                        for (int r = 0; r < 8; r++) {
                            const int qrow = qbase + r + hi * 8;
                            const int key  = klo + lo;
                            s[r] = (key <= qrow) ? s[r] * scale : -1.0e30f;
                        }
                    }
                    sc[sub] = s;
                } else {
                    v8f s;
                    #pragma unroll
                    for (int r = 0; r < 8; r++) s[r] = -1.0e30f;
                    sc[sub] = s;
                }
            }
            // --- online softmax over the 64-key chunk ---
            float sf[8];
            #pragma unroll
            for (int r = 0; r < 8; r++) {
                float mx = fmaxf(fmaxf(sc[0][r], sc[1][r]), fmaxf(sc[2][r], sc[3][r]));
                mx = rowmax16(mx);
                const float mn = fmaxf(m[r], mx);
                sf[r] = __expf(m[r] - mn);
                m[r]  = mn;
                float rs = 0.f;
                #pragma unroll
                for (int sub = 0; sub < 4; sub++) { sc[sub][r] = __expf(sc[sub][r] - mn); rs += sc[sub][r]; }
                su[r] = su[r] * sf[r] + rowsum16(rs);
            }
            #pragma unroll
            for (int c = 0; c < 8; c++)
                #pragma unroll
                for (int r = 0; r < 8; r++) acc[c][r] *= sf[r];

            // --- P: C-layout -> A-layout via per-wave LDS ---
            #pragma unroll
            for (int r = 0; r < 8; r++) {
                _Float16* pp = &Pw[(w * 16 + r + hi * 8) * SPP];
                #pragma unroll
                for (int sub = 0; sub < 4; sub++) pp[sub * 16 + lo] = (_Float16)sc[sub][r];
            }
        }
        __syncthreads();

        if (c2 < myNch) {
            // --- P (16x64) @ V (64x128): 2 K-steps x 8 n-tiles = 16 WMMAs ---
            v16h pa[2];
            #pragma unroll
            for (int kk = 0; kk < 2; kk++) {
                const _Float16* p = &Pw[(w * 16 + lo) * SPP + kk * 32 + hi * 8];
                HF u; u.h8[0] = *(const v8h*)p; u.h8[1] = *(const v8h*)(p + 16);
                pa[kk] = u.v;
            }
            #pragma unroll
            for (int c = 0; c < 8; c++) {
                #pragma unroll
                for (int kk = 0; kk < 2; kk++) {
                    const _Float16* p = &Vt[it][(c * 16 + lo) * SVV + kk * 32 + hi * 8];
                    HF u; u.h8[0] = *(const v8h*)p; u.h8[1] = *(const v8h*)(p + 16);
                    acc[c] = wmma_f16(pa[kk], u.v, acc[c]);
                }
            }
        }
        __syncthreads();   // everyone done with buffer `it` before restage
    }

    // epilogue: one reciprocal per row, then scale-store
    float rcp[8];
    #pragma unroll
    for (int r = 0; r < 8; r++) rcp[r] = __frcp_rn(su[r]);
    #pragma unroll
    for (int c = 0; c < 8; c++)
        #pragma unroll
        for (int r = 0; r < 8; r++) {
            const int row = qbase + r + hi * 8;
            ctxh[(((size_t)b * 2048 + row) * 16 + h) * 128 + c * 16 + lo] =
                (_Float16)(acc[c][r] * rcp[r]);
        }
}

// ---------------------------------------------------------------------------
extern "C" void kernel_launch(void* const* d_in, const int* in_sizes, int n_in,
                              void* d_out, int out_size, void* d_ws, size_t ws_size,
                              hipStream_t stream) {
    const float* x       = (const float*)d_in[0];   // [2,2048,2048]
    const float* w_q     = (const float*)d_in[1];   // [2048,2048]
    const float* w_kv    = (const float*)d_in[2];   // [2048,1024]
    const float* w_dense = (const float*)d_in[3];   // [2048,2048]
    float* out = (float*)d_out;

    const int M = 4096, H = 2048;

    _Float16* xh   = (_Float16*)d_ws;                  // 16 MiB
    _Float16* qh   = xh   + (size_t)M * 2048;          // 16 MiB
    _Float16* kvh  = qh   + (size_t)M * 2048;          //  8 MiB
    _Float16* ctxh = kvh  + (size_t)M * 1024;          // 16 MiB
    _Float16* wqT  = ctxh + (size_t)M * 2048;          //  8 MiB  [2048][2048]
    _Float16* wkvT = wqT  + (size_t)2048 * 2048;       //  4 MiB  [1024][2048]
    _Float16* wdT  = wkvT + (size_t)1024 * 2048;       //  8 MiB  [2048][2048]
    _Float16* vT   = wdT  + (size_t)2048 * 2048;       //  4 MiB  [8][128][2048]

    // 0) one-time converts / transposes (HBM-bound, tiny vs 23.3 TB/s)
    cvt_f16_kernel<<<(M * 2048 / 8 + 255) / 256, 256, 0, stream>>>(x, xh);
    wt_transpose_kernel<<<dim3(2048 / 32, 2048 / 32), 256, 0, stream>>>(w_q, wqT, H, 2048);
    wt_transpose_kernel<<<dim3(1024 / 32, 2048 / 32), 256, 0, stream>>>(w_kv, wkvT, H, 1024);
    wt_transpose_kernel<<<dim3(2048 / 32, 2048 / 32), 256, 0, stream>>>(w_dense, wdT, H, 2048);

    // 1) Q = x @ w_q ; 2) KV = x @ w_kv
    gemm_f16_wmma<true><<<dim3(2048 / 128, M / 128), 256, 0, stream>>>(xh, wqT, (void*)qh, M, 2048, H);
    gemm_f16_wmma<true><<<dim3(1024 / 128, M / 128), 256, 0, stream>>>(xh, wkvT, (void*)kvh, M, 1024, H);

    // 3) RoPE on Q and K
    rope_kernel<<<(2 * 2048 * 16 * 64) / 256, 256, 0, stream>>>(qh, 16, 2048, 2 * 2048 * 16 * 64);
    rope_kernel<<<(2 * 2048 * 4 * 64) / 256, 256, 0, stream>>>(kvh, 4, 1024, 2 * 2048 * 4 * 64);

    // 4) V transpose for attention B-fragments
    v_transpose_kernel<<<dim3(2048 / 32, 128 / 32, 8), 256, 0, stream>>>(kvh, vT);

    // 5) causal flash attention (8 waves/block share K/V stream)
    attn_fa_wmma<<<dim3(2048 / 32, 4, 2), 256, 0, stream>>>(qh, kvh, vT, ctxh);

    // 6) out = ctx @ w_dense (fp32 out)
    gemm_f16_wmma<false><<<dim3(2048 / 128, M / 128), 256, 0, stream>>>(ctxh, wdT, (void*)out, M, 2048, H);
}
